// TimeAttnReadout_65970697667198
// MI455X (gfx1250) — matrix operand order repaired and checked by reading, not compiled, not run on hardware
//
#include <hip/hip_runtime.h>

#define H 256

typedef __attribute__((ext_vector_type(16))) __bf16    v16bf;
typedef __attribute__((ext_vector_type(8)))  float     v8f;
typedef __attribute__((ext_vector_type(4)))  unsigned  u32x4;
typedef __attribute__((ext_vector_type(4)))  float     f32x4;
typedef __attribute__((ext_vector_type(8)))  int       i32x8;
typedef __attribute__((ext_vector_type(4)))  int       i32x4;

#ifndef __has_builtin
#define __has_builtin(x) 0
#endif
#if __has_builtin(__builtin_amdgcn_tensor_load_to_lds) && \
    __has_builtin(__builtin_amdgcn_s_wait_tensorcnt)
#define TDM_OK 1
#else
#define TDM_OK 0
#endif

union BFfrag {
    u32x4 u4[2];
    v16bf v;
};

// round-to-nearest-even f32 -> bf16, packed pair into one dword
__device__ inline unsigned pack2bf(float a, float b) {
    unsigned ua = __float_as_uint(a);
    unsigned ub = __float_as_uint(b);
    ua = (ua + 0x7FFFu + ((ua >> 16) & 1u)) >> 16;
    ub = (ub + 0x7FFFu + ((ub >> 16) & 1u)) >> 16;
    return (ub << 16) | (ua & 0xFFFFu);
}

// ---------------------------------------------------------------------------
// Kernel 0: convert the three 256x256 f32 weight matrices to bf16 (packed u32)
// ---------------------------------------------------------------------------
__global__ void wconv_kernel(const float* __restrict__ Wu,
                             const float* __restrict__ Wv,
                             const float* __restrict__ Wout,
                             unsigned* __restrict__ wu_bf,
                             unsigned* __restrict__ wv_bf,
                             unsigned* __restrict__ wout_bf) {
    const int pairs = (H * H) / 2;  // 32768 dwords per matrix
    int gid = blockIdx.x * blockDim.x + threadIdx.x;
    int mat = gid / pairs;
    int p   = gid % pairs;
    const float* src = (mat == 0) ? Wu : (mat == 1) ? Wv : Wout;
    unsigned*    dst = (mat == 0) ? wu_bf : (mat == 1) ? wv_bf : wout_bf;
    dst[p] = pack2bf(src[2 * p], src[2 * p + 1]);
}

// ---------------------------------------------------------------------------
// Kernel 1: exclusive prefix scan of ragged lengths -> segment offsets [B+1]
// ---------------------------------------------------------------------------
__global__ void scan_kernel(const int* __restrict__ lens,
                            int* __restrict__ off, int b) {
    __shared__ int partial[256];
    int t = threadIdx.x;
    int chunk = (b + 255) / 256;
    int base = t * chunk;
    int s = 0;
    for (int i = 0; i < chunk; i++) {
        int idx = base + i;
        if (idx < b) s += lens[idx];
    }
    partial[t] = s;
    __syncthreads();
    if (t == 0) {
        int run = 0;
        for (int i = 0; i < 256; i++) { int v = partial[i]; partial[i] = run; run += v; }
        off[b] = run;
    }
    __syncthreads();
    int run = partial[t];
    for (int i = 0; i < chunk; i++) {
        int idx = base + i;
        if (idx < b) { off[idx] = run; run += lens[idx]; }
    }
}

// ---------------------------------------------------------------------------
// Kernel 2: e[n] = sigmoid(feats@Wu.T + bu + ctx@Wv.T) . We   (WMMA bf16)
//   128 threads = 4 waves, 64 rows per block (16 per wave).
//   A tiles staged bf16 in LDS then hoisted into 128 VGPRs of fragments per
//   wave. Per-j-tile weight slabs (16 KB) are TDM-DMA'd into a double-buffered
//   LDS region (hardware pad -> 528B row stride), overlapped with the WMMA
//   loop. B fragments are double-buffered in registers so LDS loads pipeline
//   ahead of the matrix ops. Fallback path uses plain loads + ds stores.
// ---------------------------------------------------------------------------
#define AROWS   64
#define ASTRIDE 132                       // dwords per LDS row (128 + 4 pad)
#define ABYTES  (AROWS * ASTRIDE * 4)     // 33792 per matrix
#define BSLAB_BASE (2 * ABYTES)           // 67584
#define BMAT_BYTES (16 * ASTRIDE * 4)     // 8448  (16 rows, padded)
#define BBUF_BYTES (2 * BMAT_BYTES)       // 16896 (Wu slab + Wv slab)
#define LDS_DWORDS ((BSLAB_BASE + 2 * BBUF_BYTES) / 4)  // 25344 dw = 101376 B

#if TDM_OK
__device__ inline void issue_fill_tdm(const unsigned* wu_bf, const unsigned* wv_bf,
                                      unsigned lds_base, int jt, int buf) {
    // One 1D TDM copy per matrix: 8 KB (2048 x 4B) contiguous from global,
    // LDS pad: +4 dwords after every 128 dwords -> 528B row stride in LDS.
#pragma unroll
    for (int mat = 0; mat < 2; mat++) {
        size_t gaddr = (size_t)(mat ? wv_bf : wu_bf) + (size_t)jt * 8192;
        unsigned ldsa = lds_base + BSLAB_BASE + buf * BBUF_BYTES + mat * BMAT_BYTES;
        u32x4 g0;
        g0.x = 1u;                                        // count = 1 valid D#
        g0.y = ldsa;                                      // lds_addr
        g0.z = (unsigned)(gaddr & 0xFFFFFFFFu);           // global_addr[31:0]
        g0.w = (unsigned)((gaddr >> 32) & 0x01FFFFFFu)    // global_addr[56:32]
               | (2u << 30);                              // type = 2 (image)
        i32x8 g1;
        g1[0] = (int)((2u << 16)      // data_size = 4B
                      | (1u << 20)    // pad_enable
                      | (6u << 22)    // pad_interval: 128 dwords
                      | (3u << 25));  // pad_amount:   4 dwords
        g1[1] = (int)(2048u << 16);   // tensor_dim0[15:0]
        g1[2] = (int)(1u << 16);      // tensor_dim0[31:16]=0 | tensor_dim1 lo = 1
        g1[3] = (int)(2048u << 16);   // tensor_dim1 hi = 0 | tile_dim0 = 2048
        g1[4] = 0;                    // tile_dim1 = 0 (1D), tile_dim2 = 0
        g1[5] = 2048;                 // tensor_dim0_stride[31:0]
        g1[6] = 0;
        g1[7] = 0;
        i32x4 gz = {0, 0, 0, 0};
#if defined(__clang_major__) && __clang_major__ >= 23
        i32x8 z8 = {0, 0, 0, 0, 0, 0, 0, 0};
        __builtin_amdgcn_tensor_load_to_lds(g0, g1, gz, gz, z8, 0);
#else
        __builtin_amdgcn_tensor_load_to_lds(g0, g1, gz, gz, 0);
#endif
    }
}
#else
__device__ inline void fill_slab_fallback(unsigned* lds_u,
                                          const unsigned* __restrict__ wu_bf,
                                          const unsigned* __restrict__ wv_bf,
                                          int jt, int buf, int t) {
    for (int i = t; i < 1024; i += 128) {   // 2 matrices x 512 granules of 16B
        int mat = i >> 9;
        int idx = i & 511;
        int row = idx >> 5;
        int g   = idx & 31;
        const unsigned* src = (mat ? wv_bf : wu_bf) + (size_t)(jt * 16 + row) * 128 + g * 4;
        u32x4 v = *(const u32x4*)src;
        *(u32x4*)((unsigned char*)lds_u + BSLAB_BASE + buf * BBUF_BYTES +
                  mat * BMAT_BYTES + row * 528 + g * 16) = v;
    }
}
#endif

__global__ void __launch_bounds__(128)
score_kernel(const float* __restrict__ feats, const float* __restrict__ ctx,
             const unsigned* __restrict__ wu_bf, const unsigned* __restrict__ wv_bf,
             const float* __restrict__ bu, const float* __restrict__ We,
             float* __restrict__ e_out) {
    __shared__ unsigned lds_u[LDS_DWORDS];
    int t = threadIdx.x;                    // 0..127
    int rowbase = blockIdx.x * AROWS;

#if TDM_OK
    unsigned lds_base = (unsigned)(size_t)(void*)lds_u;
    if (t < 32) issue_fill_tdm(wu_bf, wv_bf, lds_base, 0, 0);  // prefetch jt=0
#endif

    // ---- stage 64 rows x 256 f32 of feats & ctx as bf16 into LDS ----
    for (int mtx = 0; mtx < 2; mtx++) {
        const float* src = mtx ? ctx : feats;
        unsigned* dstl = lds_u + mtx * (AROWS * ASTRIDE);
        for (int i = t; i < AROWS * 64; i += 128) {
            int row = i >> 6;
            int c4  = i & 63;
            f32x4 f = ((const f32x4*)(src + (size_t)(rowbase + row) * H))[c4];
            unsigned* d = dstl + row * ASTRIDE + c4 * 2;
            d[0] = pack2bf(f.x, f.y);
            d[1] = pack2bf(f.z, f.w);
        }
    }
#if !TDM_OK
    fill_slab_fallback(lds_u, wu_bf, wv_bf, 0, 0, t);
#endif
    __syncthreads();   // A tiles staged & visible

    int wave = t >> 5;
    int lane = t & 31;
    int half = lane >> 4;
    int m    = lane & 15;

    const unsigned char* lbase = (const unsigned char*)lds_u;
    const int rowb = (wave * 16 + m) * (ASTRIDE * 4);

    // ---- hoist the wave's A fragments (j-tile invariant) into registers ----
    // 16 fragments x 8 VGPRs = 128 VGPRs; hot loop then only streams B slabs.
    BFfrag afr[8], acr[8];
#pragma unroll
    for (int kc = 0; kc < 8; kc++) {
        int k0 = kc * 32;
        int o1 = rowb + (k0 + half * 16) * 2;       // contiguous k = 16*half + s
        int o2 = o1 + 16;
        afr[kc].u4[0] = *(const u32x4*)(lbase + o1);
        afr[kc].u4[1] = *(const u32x4*)(lbase + o2);
        acr[kc].u4[0] = *(const u32x4*)(lbase + ABYTES + o1);
        acr[kc].u4[1] = *(const u32x4*)(lbase + ABYTES + o2);
    }

    float eacc[8];
#pragma unroll
    for (int r = 0; r < 8; r++) eacc[r] = 0.f;

    for (int jt = 0; jt < 16; jt++) {
        int buf = jt & 1;
#if TDM_OK
        if (t < 32) __builtin_amdgcn_s_wait_tensorcnt(0);  // slab jt landed
#endif
        __syncthreads();   // slab visible; everyone done with other buffer
#if TDM_OK
        if (t < 32 && jt + 1 < 16)
            issue_fill_tdm(wu_bf, wv_bf, lds_base, jt + 1, buf ^ 1);
#else
        if (jt + 1 < 16) fill_slab_fallback(lds_u, wu_bf, wv_bf, jt + 1, buf ^ 1, t);
#endif

        int j = jt * 16 + m;
        float bj = bu[j];
        v8f c;
#pragma unroll
        for (int r = 0; r < 8; r++) c[r] = bj;  // bias via C init

        const unsigned char* bb = lbase + BSLAB_BASE + buf * BBUF_BYTES + m * 528;

        // B fragments double-buffered in registers: loads for chunk kc+1 are
        // issued with independent destinations so they pipeline past the
        // WMMAs of chunk kc (avoids s_wait_dscnt 0 before every WMMA).
        BFfrag bwu[2], bwv[2];
        {
            int bofs = half * 32;  // kc = 0
            bwu[0].u4[0] = *(const u32x4*)(bb + bofs);
            bwu[0].u4[1] = *(const u32x4*)(bb + bofs + 16);
            bwv[0].u4[0] = *(const u32x4*)(bb + BMAT_BYTES + bofs);
            bwv[0].u4[1] = *(const u32x4*)(bb + BMAT_BYTES + bofs + 16);
        }
#pragma unroll
        for (int kc = 0; kc < 8; kc++) {
            int cur = kc & 1, nxt = cur ^ 1;
            if (kc + 1 < 8) {
                int bofs = (kc + 1) * 64 + half * 32;
                bwu[nxt].u4[0] = *(const u32x4*)(bb + bofs);
                bwu[nxt].u4[1] = *(const u32x4*)(bb + bofs + 16);
                bwv[nxt].u4[0] = *(const u32x4*)(bb + BMAT_BYTES + bofs);
                bwv[nxt].u4[1] = *(const u32x4*)(bb + BMAT_BYTES + bofs + 16);
            }
            c = __builtin_amdgcn_wmma_f32_16x16x32_bf16(false, afr[kc].v, false, bwu[cur].v,
                                                        (short)0, c, false, false);
            c = __builtin_amdgcn_wmma_f32_16x16x32_bf16(false, acr[kc].v, false, bwv[cur].v,
                                                        (short)0, c, false, false);
        }

        // sigmoid, scale by We[j], reduce across the 16 columns of the tile
        float wj = We[j];
#pragma unroll
        for (int r = 0; r < 8; r++) {
            float s = wj / (1.0f + __expf(-c[r]));
            s += __shfl_xor(s, 1, 32);
            s += __shfl_xor(s, 2, 32);
            s += __shfl_xor(s, 4, 32);
            s += __shfl_xor(s, 8, 32);
            eacc[r] += s;
        }
    }

    // C layout: VGPR r holds M=r (lanes 0-15) and M=8+r (lanes 16-31)
    if (m == 0) {
        int gr = rowbase + wave * 16 + half * 8;
#pragma unroll
        for (int r = 0; r < 8; r++) e_out[gr + r] = eacc[r];
    }
}

// ---------------------------------------------------------------------------
// Kernel 3: per-segment softmax over e, weighted pooling of feats -> rst bf16
// ---------------------------------------------------------------------------
__global__ void __launch_bounds__(256)
pool_kernel(const float* __restrict__ feats, const float* __restrict__ e,
            const int* __restrict__ off, unsigned short* __restrict__ rstb) {
    __shared__ float red[256];
    __shared__ float alpha[512];
    int b = blockIdx.x;
    int t = threadIdx.x;
    int s0 = off[b];
    int len = off[b + 1] - s0;

    float mx = -3.4e38f;
    for (int i = t; i < len; i += 256) mx = fmaxf(mx, e[s0 + i]);
    red[t] = mx;
    __syncthreads();
    for (int o = 128; o > 0; o >>= 1) {
        if (t < o) red[t] = fmaxf(red[t], red[t + o]);
        __syncthreads();
    }
    mx = red[0];
    __syncthreads();

    float sm = 0.f;
    for (int i = t; i < len; i += 256) {
        float ex = __expf(e[s0 + i] - mx);
        if (i < 512) alpha[i] = ex;
        sm += ex;
    }
    red[t] = sm;
    __syncthreads();
    for (int o = 128; o > 0; o >>= 1) {
        if (t < o) red[t] += red[t + o];
        __syncthreads();
    }
    float inv = 1.0f / red[0];

    float acc = 0.f;
    for (int i = 0; i < len; i++)
        acc += alpha[i] * feats[(size_t)(s0 + i) * H + t];
    acc *= inv;

    unsigned ua = __float_as_uint(acc);
    ua = (ua + 0x7FFFu + ((ua >> 16) & 1u)) >> 16;
    rstb[(size_t)b * H + t] = (unsigned short)ua;
}

// ---------------------------------------------------------------------------
// Kernel 4: out = rst(bf16) @ Wout.T(bf16) via WMMA, f32 output
// ---------------------------------------------------------------------------
__global__ void __launch_bounds__(128)
out_kernel(const unsigned char* __restrict__ rstb,
           const unsigned char* __restrict__ wout_bf,
           float* __restrict__ out) {
    int t = threadIdx.x;
    int wave = t >> 5, lane = t & 31;
    int half = lane >> 4, m = lane & 15;
    int base = blockIdx.x * 64 + wave * 16;

    for (int jt = 0; jt < 16; jt++) {
        v8f c;
#pragma unroll
        for (int r = 0; r < 8; r++) c[r] = 0.f;

#pragma unroll
        for (int kc = 0; kc < 8; kc++) {
            int k0 = kc * 32;
            BFfrag a, bm;
            const unsigned char* arow = rstb + (size_t)(base + m) * (H * 2);
            a.u4[0] = *(const u32x4*)(arow + (k0 + half * 16) * 2);
            a.u4[1] = *(const u32x4*)(arow + (k0 + half * 16) * 2 + 16);
            const unsigned char* brow = wout_bf + (size_t)(jt * 16 + m) * (H * 2);
            bm.u4[0] = *(const u32x4*)(brow + (k0 + half * 16) * 2);
            bm.u4[1] = *(const u32x4*)(brow + (k0 + half * 16) * 2 + 16);
            c = __builtin_amdgcn_wmma_f32_16x16x32_bf16(false, a.v, false, bm.v,
                                                        (short)0, c, false, false);
        }
#pragma unroll
        for (int r = 0; r < 8; r++)
            out[(size_t)(base + half * 8 + r) * H + jt * 16 + m] = c[r];
    }
}

// ---------------------------------------------------------------------------
extern "C" void kernel_launch(void* const* d_in, const int* in_sizes, int n_in,
                              void* d_out, int out_size, void* d_ws, size_t ws_size,
                              hipStream_t stream) {
    const float* feats = (const float*)d_in[0];
    const float* ctx   = (const float*)d_in[1];
    const int*   lens  = (const int*)d_in[2];
    const float* Wu    = (const float*)d_in[3];
    const float* bu    = (const float*)d_in[4];
    const float* Wv    = (const float*)d_in[5];
    const float* We    = (const float*)d_in[6];
    const float* Wout  = (const float*)d_in[7];
    float* out = (float*)d_out;

    const size_t N = (size_t)in_sizes[0] / H;  // 204800
    const int    B = in_sizes[2];              // 4096

    auto al16 = [](size_t x) { return (x + 15) & ~(size_t)15; };
    char* ws = (char*)d_ws;
    size_t o = 0;
    float* e = (float*)(ws + o);              o = al16(o + N * sizeof(float));
    int* off = (int*)(ws + o);                o = al16(o + (size_t)(B + 1) * sizeof(int));
    unsigned short* rstb = (unsigned short*)(ws + o);
                                              o = al16(o + (size_t)B * H * sizeof(unsigned short));
    unsigned* wu_bf = (unsigned*)(ws + o);    o = al16(o + (size_t)H * H / 2 * sizeof(unsigned));
    unsigned* wv_bf = (unsigned*)(ws + o);    o = al16(o + (size_t)H * H / 2 * sizeof(unsigned));
    unsigned* wout_bf = (unsigned*)(ws + o);  o = al16(o + (size_t)H * H / 2 * sizeof(unsigned));
    (void)ws_size; (void)n_in; (void)out_size;

    wconv_kernel<<<(3 * (H * H / 2)) / 256, 256, 0, stream>>>(Wu, Wv, Wout,
                                                              wu_bf, wv_bf, wout_bf);
    scan_kernel<<<1, 256, 0, stream>>>(lens, off, B);
    score_kernel<<<(unsigned)(N / AROWS), 128, 0, stream>>>(feats, ctx, wu_bf, wv_bf,
                                                            bu, We, e);
    pool_kernel<<<B, 256, 0, stream>>>(feats, e, off, rstb);
    out_kernel<<<B / 64, 128, 0, stream>>>((const unsigned char*)rstb,
                                           (const unsigned char*)wout_bf, out);
}
